// TestModel_38800734552681
// MI455X (gfx1250) — matrix operand order repaired and checked by reading, not compiled, not run on hardware
//
#include <hip/hip_runtime.h>
#include <stdint.h>

// ---------------------------------------------------------------------------
// Types for CDNA5 WMMA (wave32): 16x16x32 bf16 -> f32
// ---------------------------------------------------------------------------
typedef __attribute__((ext_vector_type(16))) __bf16   v16bf;
typedef __attribute__((ext_vector_type(8)))  __bf16   v8bf;
typedef __attribute__((ext_vector_type(8)))  float    v8f;
typedef __attribute__((ext_vector_type(4)))  uint32_t u32x4;
typedef __attribute__((ext_vector_type(8)))  uint32_t u32x8;

#define D_   256      // output features
#define K_   512      // input features (2D)
#define MT   16       // rows per workgroup tile
#define LDA  520      // ushort stride of bf16 A staging (1040 B rows)
#define LDY  260      // float stride of Y buffer
#define LDF  260      // float stride of TDM-staged f32 halves (256 + 4 pad dwords)

// RNE f32 -> bf16 bits
__device__ __forceinline__ unsigned short f2bf(float f) {
  union { float f; uint32_t u; } c; c.f = f;
  uint32_t u = c.u;
  return (unsigned short)((u + 0x7FFFu + ((u >> 16) & 1u)) >> 16);
}

// ---------------------------------------------------------------------------
// Tensor Data Mover: DMA a [rows x 256] f32 tile (row stride = stride_elems)
// from global memory into LDS at lds_off, padding each 256-dword row by 4
// dwords (LDS row stride 260 dwords).  Rows beyond tensor_dim1 are zero-filled.
// D# layout per CDNA5 ISA (2D tensor, groups 2/3
// zeroed).  Issued per-wave (EXEC ignored); tracked by TENSORcnt.
// ---------------------------------------------------------------------------
__device__ __forceinline__ void tdm_load_2d(uint32_t lds_off, const float* gptr,
                                            uint32_t rows, uint32_t stride_elems) {
  const uint64_t ga = (uint64_t)(uintptr_t)gptr;
  u32x4 g0;
  g0[0] = 1u;                                               // count=1, no gather
  g0[1] = lds_off;                                          // lds_addr (bytes)
  g0[2] = (uint32_t)ga;                                     // global_addr[31:0]
  g0[3] = (uint32_t)((ga >> 32) & 0x01FFFFFFu) | (2u << 30); // addr[56:32] | type=2
  u32x8 g1;
  g1[0] = (2u << 16)            // data_size = 4 B
        | (1u << 20)            // pad_enable
        | (7u << 22)            // pad_interval: every 256 dwords
        | (3u << 25);           // pad_amount: 4 dwords
  g1[1] = (stride_elems & 0xFFFFu) << 16;                   // tensor_dim0[15:0]
  g1[2] = ((stride_elems >> 16) & 0xFFFFu)                  // tensor_dim0[31:16]
        | ((rows & 0xFFFFu) << 16);                         // tensor_dim1[15:0]
  g1[3] = ((rows >> 16) & 0xFFFFu)                          // tensor_dim1[31:16]
        | (256u << 16);                                     // tile_dim0 = 256
  g1[4] = (uint32_t)MT;                                     // tile_dim1=16, tile_dim2=0
  g1[5] = stride_elems;                                     // tensor_dim0_stride[31:0]
  g1[6] = 0u;                                               // stride0[47:32], stride1 lo
  g1[7] = 0u;
  u32x4 z4 = {0u, 0u, 0u, 0u};
  asm volatile("tensor_load_to_lds %0, %1, %2, %3"
               :: "s"(g0), "s"(g1), "s"(z4), "s"(z4)
               : "memory");
}

// A fragment (16x32 bf16) from LDS, documented 16-bit A layout:
// lanes 0-15: elems 0-7 = K[k0..k0+7],    elems 8-15 = K[k0+16..k0+23]
// lanes16-31: elems 0-7 = K[k0+8..k0+15], elems 8-15 = K[k0+24..k0+31]
__device__ __forceinline__ v16bf load_frag_a(const unsigned short* base, int lane, int k0) {
  const int row  = lane & 15;
  const int koff = k0 + ((lane >> 4) << 3);
  const __bf16* p = reinterpret_cast<const __bf16*>(base + row * LDA + koff);
  v8bf lo = *reinterpret_cast<const v8bf*>(p);
  v8bf hi = *reinterpret_cast<const v8bf*>(p + 16);
  return __builtin_shufflevector(lo, hi, 0,1,2,3,4,5,6,7,8,9,10,11,12,13,14,15);
}

// B fragment (32x16 bf16) from global bf16 W[N][K] row-major, dense-B layout:
// lane%16 = column N; lanes 0-15 hold K[k0..k0+7],K[k0+16..k0+23]; lanes 16-31 hold K[k0+8..k0+15],K[k0+24..k0+31]
__device__ __forceinline__ v16bf load_frag_b(const unsigned short* W, int lane, int nb, int k0) {
  const int col  = nb + (lane & 15);
  const int koff = k0 + ((lane >> 4) << 3);
  const __bf16* p = reinterpret_cast<const __bf16*>(W + (size_t)col * K_ + koff);
  v8bf lo = *reinterpret_cast<const v8bf*>(p);
  v8bf hi = *reinterpret_cast<const v8bf*>(p + 16);
  return __builtin_shufflevector(lo, hi, 0,1,2,3,4,5,6,7,8,9,10,11,12,13,14,15);
}

// ---------------------------------------------------------------------------
// Utility kernels
// ---------------------------------------------------------------------------
__global__ void f32_to_bf16_kernel(const float* __restrict__ src,
                                   unsigned short* __restrict__ dst, int n) {
  int i = blockIdx.x * blockDim.x + threadIdx.x;
  if (i < n) dst[i] = f2bf(src[i]);
}

__global__ void zero_kernel(float* __restrict__ p, int n) {
  int i = blockIdx.x * blockDim.x + threadIdx.x;
  if (i < n) p[i] = 0.0f;
}

// ---------------------------------------------------------------------------
// Fused level block:  Y = LIF(RMSNorm(concat(A0,A1) @ W^T + b)) + residual
//   mode 0 (up):   res = 0.5*(A0row+A1row); out[r]     = block + res
//   mode 1 (down): res = A0row (ctx);       out[2r]    = res
//                                           out[2r+1]  = block + res
//   mode 2 (leaf): out[r] = block
// Block: 256 threads = 8 wave32; tile = 16 rows x 256 cols; K = 512.
// Inputs staged via TDM (f32), converted once to bf16 for the WMMA loop.
// ---------------------------------------------------------------------------
__global__ __launch_bounds__(256) void fused_level_kernel(
    const float* __restrict__ A0, long long sa0,
    const float* __restrict__ A1, long long sa1,
    const unsigned short* __restrict__ Wb,
    const float* __restrict__ bias,
    const float* __restrict__ nw,
    const float* __restrict__ thr,
    float* __restrict__ Y,
    int M, int mode)
{
  __shared__ float Af0[MT * LDF];            // TDM-staged f32 half 0 (16,640 B)
  __shared__ float Af1[MT * LDF];            // TDM-staged f32 half 1 (16,640 B)
  __shared__ char  UBuf[MT * LDA * 2];       // bf16 A (GEMM phase) / f32 Y (epilogue)
  __shared__ float red[MT * 16];
  __shared__ float invr[MT];
  __shared__ float den[MT];
  unsigned short* Abuf = reinterpret_cast<unsigned short*>(UBuf);
  float*          Ybuf = reinterpret_cast<float*>(UBuf);

  const int t    = threadIdx.x;
  const int lane = t & 31;
  const int w    = t >> 5;

  const int rr   = t >> 4;           // 0..15 : row within tile
  const int cseg = (t & 15) << 4;    // 16-col segment
  const long long gr = (long long)blockIdx.x * MT + rr;
  const bool valid = (gr < M);

  // ---- TDM: DMA both f32 halves of the A tile into LDS (wave 0 issues) ----
  if (t == 0) {
    const long long r0 = (long long)blockIdx.x * MT;
    long long rem = (long long)M - r0;
    uint32_t rows = (uint32_t)(rem < MT ? rem : MT);   // OOB rows zero-filled
    tdm_load_2d((uint32_t)(uintptr_t)(void*)Af0, A0 + r0 * sa0, rows, (uint32_t)sa0);
    tdm_load_2d((uint32_t)(uintptr_t)(void*)Af1, A1 + r0 * sa1, rows, (uint32_t)sa1);
    __builtin_amdgcn_s_wait_tensorcnt(0);
  }
  __syncthreads();

  // ---- convert staged f32 -> bf16 fragment buffer (packed dword stores) ----
  {
    #pragma unroll
    for (int i = 0; i < 16; i += 4) {
      const int c = cseg + i;
      float4 v0 = *reinterpret_cast<const float4*>(&Af0[rr * LDF + c]);
      float4 v1 = *reinterpret_cast<const float4*>(&Af1[rr * LDF + c]);
      uint2 p0, p1;
      p0.x = (uint32_t)f2bf(v0.x) | ((uint32_t)f2bf(v0.y) << 16);
      p0.y = (uint32_t)f2bf(v0.z) | ((uint32_t)f2bf(v0.w) << 16);
      p1.x = (uint32_t)f2bf(v1.x) | ((uint32_t)f2bf(v1.y) << 16);
      p1.y = (uint32_t)f2bf(v1.z) | ((uint32_t)f2bf(v1.w) << 16);
      *reinterpret_cast<uint2*>(&Abuf[rr * LDA + c])       = p0;
      *reinterpret_cast<uint2*>(&Abuf[rr * LDA + 256 + c]) = p1;
    }
  }
  __syncthreads();

  // ---- GEMM: wave w -> rows 0..15, cols [w*32, w*32+32) ----
  v8f acc0 = {};
  v8f acc1 = {};
  const int nb0 = w << 5;
  #pragma unroll
  for (int k0 = 0; k0 < K_; k0 += 32) {
    v16bf a  = load_frag_a(Abuf, lane, k0);
    v16bf b0 = load_frag_b(Wb, lane, nb0, k0);
    v16bf b1 = load_frag_b(Wb, lane, nb0 + 16, k0);
    acc0 = __builtin_amdgcn_wmma_f32_16x16x32_bf16(false, a, false, b0,
                                                   (short)0, acc0, false, false);
    acc1 = __builtin_amdgcn_wmma_f32_16x16x32_bf16(false, a, false, b1,
                                                   (short)0, acc1, false, false);
  }
  __syncthreads();   // Abuf dead; Ybuf aliases it

  // ---- spill accumulators (+bias) to LDS.  C layout: lane%16 = N,
  //      rows = elem + (lane>=16 ? 8 : 0) ----
  {
    const int roff = (lane >> 4) ? 8 : 0;
    const int c0 = nb0 + (lane & 15);
    const int c1 = nb0 + 16 + (lane & 15);
    const float b0v = bias[c0];
    const float b1v = bias[c1];
    #pragma unroll
    for (int j = 0; j < 8; ++j) {
      Ybuf[(roff + j) * LDY + c0] = acc0[j] + b0v;
      Ybuf[(roff + j) * LDY + c1] = acc1[j] + b1v;
    }
  }
  __syncthreads();

  // ---- RMSNorm reduction: mean of squares per row ----
  {
    float ss = 0.f;
    #pragma unroll
    for (int i = 0; i < 16; ++i) {
      float y = Ybuf[rr * LDY + cseg + i];
      ss += y * y;
    }
    red[rr * 16 + (t & 15)] = ss;
  }
  __syncthreads();
  if (t < MT) {
    float s = 0.f;
    #pragma unroll
    for (int i = 0; i < 16; ++i) s += red[t * 16 + i];
    invr[t] = rsqrtf(s * (1.0f / 256.0f) + 1.1920929e-07f);
  }
  __syncthreads();

  // ---- spike count per row ----
  const float iv = invr[rr];
  {
    float cnt = 0.f;
    #pragma unroll
    for (int i = 0; i < 16; ++i) {
      const int c = cseg + i;
      const float v = Ybuf[rr * LDY + c] * iv * nw[c];
      cnt += (v >= thr[c]) ? 1.f : 0.f;
    }
    red[rr * 16 + (t & 15)] = cnt;
  }
  __syncthreads();
  if (t < MT) {
    float s = 0.f;
    #pragma unroll
    for (int i = 0; i < 16; ++i) s += red[t * 16 + i];
    den[t] = (s >= 1.f) ? s : 1.f;
  }
  __syncthreads();

  // ---- epilogue: spike/denom + residual (exact f32), write out ----
  if (valid) {
    const float rdn = 1.0f / den[rr];
    #pragma unroll
    for (int i = 0; i < 16; ++i) {
      const int c = cseg + i;
      const float v = Ybuf[rr * LDY + c] * iv * nw[c];
      const float o = (v >= thr[c]) ? rdn : 0.f;
      float res;
      if (mode == 0)
        res = 0.5f * (Af0[rr * LDF + c] + Af1[rr * LDF + c]);
      else if (mode == 1)
        res = Af0[rr * LDF + c];
      else
        res = 0.f;
      if (mode == 1) {
        Y[(size_t)(2 * gr) * D_ + c]     = res;       // ctx copy (even)
        Y[(size_t)(2 * gr + 1) * D_ + c] = o + res;   // ctx_right (odd)
      } else {
        Y[(size_t)gr * D_ + c] = o + res;
      }
    }
  }
}

// ---------------------------------------------------------------------------
// Host orchestration
// ---------------------------------------------------------------------------
extern "C" void kernel_launch(void* const* d_in, const int* in_sizes, int n_in,
                              void* d_out, int out_size, void* d_ws, size_t ws_size,
                              hipStream_t stream) {
  (void)in_sizes; (void)n_in; (void)out_size; (void)ws_size;
  const int B = 16, T = 8192, NL = 13;

  const float* x        = (const float*)d_in[0];
  const float* up_W     = (const float*)d_in[1];
  const float* up_b     = (const float*)d_in[2];
  const float* up_nw    = (const float*)d_in[3];
  const float* up_thr   = (const float*)d_in[4];
  const float* down_W   = (const float*)d_in[5];
  const float* down_b   = (const float*)d_in[6];
  const float* down_nw  = (const float*)d_in[7];
  const float* down_thr = (const float*)d_in[8];
  const float* leaf_W   = (const float*)d_in[9];
  const float* leaf_b   = (const float*)d_in[10];
  const float* leaf_nw  = (const float*)d_in[11];
  const float* leaf_thr = (const float*)d_in[12];

  char* ws = (char*)d_ws;
  size_t off = 0;
  const size_t WMAT = (size_t)D_ * K_;           // 131072 elems per matrix

  unsigned short* Wup = (unsigned short*)(ws + off); off += 4 * WMAT * 2;
  unsigned short* Wdn = (unsigned short*)(ws + off); off += 4 * WMAT * 2;
  unsigned short* Wlf = (unsigned short*)(ws + off); off += 1 * WMAT * 2;
  off = (off + 255) & ~(size_t)255;

  float* summ[NL + 1];
  summ[0] = nullptr;  // level 0 is x itself
  for (int l = 1; l <= NL; ++l) {
    summ[l] = (float*)(ws + off);
    off += (size_t)B * (T >> l) * D_ * 4;
    off = (off + 255) & ~(size_t)255;
  }
  float* ctxA = (float*)(ws + off); off += (size_t)B * T * D_ * 4;
  float* ctxB = (float*)(ws + off); off += (size_t)B * T * D_ * 4;

  // --- weight conversion to bf16 (L2-resident thereafter) ---
  {
    int n = (int)(4 * WMAT);
    f32_to_bf16_kernel<<<(n + 255) / 256, 256, 0, stream>>>(up_W, Wup, n);
    f32_to_bf16_kernel<<<(n + 255) / 256, 256, 0, stream>>>(down_W, Wdn, n);
    int nl = (int)WMAT;
    f32_to_bf16_kernel<<<(nl + 255) / 256, 256, 0, stream>>>(leaf_W, Wlf, nl);
  }

  // --- up-sweep: 13 fused level blocks ---
  const float* curr = x;
  for (int l = 0; l < NL; ++l) {
    const int M  = B * (T >> (l + 1));
    const int wi = (l < 3) ? l : 3;
    fused_level_kernel<<<(M + MT - 1) / MT, 256, 0, stream>>>(
        curr, 512, curr + 256, 512,
        Wup + (size_t)wi * WMAT, up_b + wi * D_, up_nw + wi * D_, up_thr + wi * D_,
        summ[l + 1], M, 0);
    curr = summ[l + 1];
  }

  // --- down-sweep ---
  {
    int n = B * D_;
    zero_kernel<<<(n + 255) / 256, 256, 0, stream>>>(ctxA, n);
  }
  float* ctx = ctxA;
  float* nxt = ctxB;
  for (int it = 0; it < NL; ++it) {
    const int l  = NL - it;
    const int M  = B * (1 << it);
    const int wi = ((l - 1) < 3) ? (l - 1) : 3;
    const float* child = (l - 1 == 0) ? x : summ[l - 1];
    fused_level_kernel<<<(M + MT - 1) / MT, 256, 0, stream>>>(
        ctx, 256, child, 512,
        Wdn + (size_t)wi * WMAT, down_b + wi * D_, down_nw + wi * D_, down_thr + wi * D_,
        nxt, M, 1);
    float* tmp = ctx; ctx = nxt; nxt = tmp;
  }

  // --- leaf block -> d_out ---
  {
    const int M = B * T;
    fused_level_kernel<<<(M + MT - 1) / MT, 256, 0, stream>>>(
        x, 256, ctx, 256,
        Wlf, leaf_b, leaf_nw, leaf_thr,
        (float*)d_out, M, 2);
  }
}
